// MyCausalMHA_8899172237936
// MI455X (gfx1250) — matrix operand order repaired and checked by reading, not compiled, not run on hardware
//
#include <hip/hip_runtime.h>

typedef __attribute__((ext_vector_type(16))) _Float16 v16h;
typedef __attribute__((ext_vector_type(8)))  _Float16 v8h;
typedef __attribute__((ext_vector_type(8)))  float    v8f;
typedef __attribute__((ext_vector_type(4)))  float    v4f;
typedef __attribute__((ext_vector_type(4)))  int      v4i;

#define D_MODEL 1024
#define N_HEADS 16
#define D_K     64
#define SEQ     2048
#define BATCH   4

// GEMM tiling
#define MT 128
#define NT 128
#define KT 32

// ---------------------------------------------------------------------------
// gfx1250 async global->LDS copy (ASYNCcnt-tracked); signature probe-confirmed:
//   (v4i __device__*, v4i __shared__*, Ii offset, Ii cpol)
// ---------------------------------------------------------------------------
#if defined(__has_builtin)
#if __has_builtin(__builtin_amdgcn_global_load_async_to_lds_b128)
#define HAVE_ASYNC_LDS 1
#endif
#endif

#ifdef HAVE_ASYNC_LDS
typedef __attribute__((address_space(1))) v4i* gas_v4i_p;
typedef __attribute__((address_space(3))) v4i* las_v4i_p;
// copy 32 contiguous bytes (16 f16) per lane: 2x b128 with matching imm offsets
#define ASYNC_CP32(gp, lp) do {                                              \
    gas_v4i_p g_ = (gas_v4i_p)(gp);                                          \
    las_v4i_p l_ = (las_v4i_p)(lp);                                          \
    __builtin_amdgcn_global_load_async_to_lds_b128(g_, l_, 0, 0);            \
    __builtin_amdgcn_global_load_async_to_lds_b128(g_, l_, 16, 0);           \
  } while (0)

static __device__ __forceinline__ void wait_async0() {
#if __has_builtin(__builtin_amdgcn_s_wait_asynccnt)
  __builtin_amdgcn_s_wait_asynccnt(0);
#else
  asm volatile("s_wait_asynccnt 0" ::: "memory");
#endif
}
#endif

// ---------------------------------------------------------------------------
// Fragment helpers for v_wmma_f32_16x16x32_f16 (wave32, layouts per ISA 7.12.2)
//   A (16x32, MxK): lane<16 -> M=lane,   K = {0..7, 16..23}
//                   lane>=16-> M=lane-16,K = {8..15,24..31}
//   B (32x16, KxN): lane<16 -> N=lane,   K = {0..7, 16..23}
//                   lane>=16-> N=lane-16,K = {8..15,24..31}
//   C/D (16x16 f32): VGPR j -> M = j + (lane<16?0:8), N = lane&15
// ---------------------------------------------------------------------------
static __device__ __forceinline__ v16h combine16(v8h lo, v8h hi) {
  v16h r;
#pragma unroll
  for (int i = 0; i < 8; ++i) { r[i] = lo[i]; r[i + 8] = hi[i]; }
  return r;
}

// 16 f16 as two 8-element runs: p[0..7] and p[16..23] (two b128 loads)
static __device__ __forceinline__ v16h frag16_lds(const _Float16* p) {
  return combine16(*(const v8h*)p, *(const v8h*)(p + 16));
}

// A fragment from f16 row (two 8-element runs: kb.. and 16+kb..)
static __device__ __forceinline__ v16h frag_a_lds(const _Float16* row, int kb) {
  return combine16(*(const v8h*)(row + kb), *(const v8h*)(row + 16 + kb));
}

// 16 f16 from two 8-float f32 runs, scaled (used for Q fragments)
static __device__ __forceinline__ v16h frag16_from_f32_scl(const float* p0, const float* p1, float scl) {
  const v4f* a = (const v4f*)p0;
  const v4f* b = (const v4f*)p1;
  v4f x0 = a[0], x1 = a[1], x2 = b[0], x3 = b[1];
  v16h r;
#pragma unroll
  for (int i = 0; i < 4; ++i) {
    r[i]      = (_Float16)(x0[i] * scl);
    r[i + 4]  = (_Float16)(x1[i] * scl);
    r[i + 8]  = (_Float16)(x2[i] * scl);
    r[i + 12] = (_Float16)(x3[i] * scl);
  }
  return r;
}

// ---------------------------------------------------------------------------
// One-pass f32 -> f16 conversion (8 elements/thread); n must be multiple of 8.
// ---------------------------------------------------------------------------
__global__ void cvt_f32_f16(const float* __restrict__ in, _Float16* __restrict__ outp, int n) {
  const int i = (blockIdx.x * blockDim.x + threadIdx.x) * 8;
  if (i >= n) return;
  v4f a = *(const v4f*)(in + i);
  v4f b = *(const v4f*)(in + i + 4);
  v8h r;
#pragma unroll
  for (int j = 0; j < 4; ++j) { r[j] = (_Float16)a[j]; r[j + 4] = (_Float16)b[j]; }
  *(v8h*)(outp + i) = r;
}

// ---------------------------------------------------------------------------
// out[m][n] = sum_k X[m][k]*W[n][k] + bias[n]   (X, W pre-converted to f16)
// 256 threads = 8 waves; block tile 128x128; wave tile 32x64; K-step 32.
// Double-buffered f16 LDS tiles; async global->LDS copies when available.
// Inner loop: ds_load_b128 + v_wmma only (no converts).
// ---------------------------------------------------------------------------
template <typename OutT>
__global__ __launch_bounds__(256) void gemm_h(const _Float16* __restrict__ X,
                                              const _Float16* __restrict__ W,
                                              const float* __restrict__ bias,
                                              OutT* __restrict__ out,
                                              int M, int N, int K) {
  __shared__ alignas(16) _Float16 Xs[2][MT][KT];   // 16 KB
  __shared__ alignas(16) _Float16 Ws[2][NT][KT];   // 16 KB

  const int tid    = threadIdx.x;
  const int lane   = tid & 31;
  const int wave   = tid >> 5;          // 0..7
  const int wm     = wave >> 1;         // 0..3 : 32-row strip
  const int wn     = wave & 1;          // 0..1 : 64-col strip
  const int colL   = lane & 15;
  const int kbA    = (lane < 16) ? 0 : 8;
  const int kbB    = (lane < 16) ? 0 : 8;
  const int rowoff = (lane < 16) ? 0 : 8;

  const int m0 = blockIdx.y * MT;
  const int n0 = blockIdx.x * NT;

  // staging: each thread owns one 16-half (32 B) segment of X and of W
  const int srow = tid >> 1;            // 0..127
  const int sseg = (tid & 1) * 16;      // 0 or 16 (halfs)
  const _Float16* gX = X + (size_t)(m0 + srow) * K + sseg;
  const _Float16* gW = W + (size_t)(n0 + srow) * K + sseg;

  v8f zero = {};
  v8f acc[2][4] = {{zero, zero, zero, zero}, {zero, zero, zero, zero}};

  const int KSTEPS = K / KT;

  // prologue: stage tile 0 into buffer 0
#ifdef HAVE_ASYNC_LDS
  ASYNC_CP32(gX, &Xs[0][srow][sseg]);
  ASYNC_CP32(gW, &Ws[0][srow][sseg]);
  wait_async0();
#else
  {
    const v4i* px = (const v4i*)gX;
    const v4i* pw = (const v4i*)gW;
    v4i* qx = (v4i*)&Xs[0][srow][sseg];
    v4i* qw = (v4i*)&Ws[0][srow][sseg];
#pragma unroll
    for (int i = 0; i < 2; ++i) { qx[i] = px[i]; qw[i] = pw[i]; }
  }
#endif
  __syncthreads();

  for (int ks = 0; ks < KSTEPS; ++ks) {
    const int cur  = ks & 1;
    const bool more = (ks + 1) < KSTEPS;

    // issue next tile's global traffic before the math
#ifdef HAVE_ASYNC_LDS
    if (more) {
      ASYNC_CP32(gX + (ks + 1) * KT, &Xs[cur ^ 1][srow][sseg]);
      ASYNC_CP32(gW + (ks + 1) * KT, &Ws[cur ^ 1][srow][sseg]);
    }
#else
    v4i rx[2], rw[2];
    if (more) {
      const v4i* px = (const v4i*)(gX + (ks + 1) * KT);
      const v4i* pw = (const v4i*)(gW + (ks + 1) * KT);
#pragma unroll
      for (int i = 0; i < 2; ++i) { rx[i] = px[i]; rw[i] = pw[i]; }
    }
#endif

    // compute: 2 A frags x 4 B frags -> 8 WMMAs (pure ds_load + wmma)
    v16h af[2], bf[4];
#pragma unroll
    for (int sm = 0; sm < 2; ++sm)
      af[sm] = frag_a_lds(&Xs[cur][wm * 32 + sm * 16 + colL][0], kbA);
#pragma unroll
    for (int sn = 0; sn < 4; ++sn)
      bf[sn] = frag16_lds(&Ws[cur][wn * 64 + sn * 16 + colL][kbB]);
#pragma unroll
    for (int sm = 0; sm < 2; ++sm)
#pragma unroll
      for (int sn = 0; sn < 4; ++sn)
        acc[sm][sn] = __builtin_amdgcn_wmma_f32_16x16x32_f16(
            false, af[sm], false, bf[sn], (short)0, acc[sm][sn], false, false);

#ifdef HAVE_ASYNC_LDS
    wait_async0();          // drain prefetch (overlapped with the WMMAs above)
#else
    if (more) {
      v4i* qx = (v4i*)&Xs[cur ^ 1][srow][sseg];
      v4i* qw = (v4i*)&Ws[cur ^ 1][srow][sseg];
#pragma unroll
      for (int i = 0; i < 2; ++i) { qx[i] = rx[i]; qw[i] = rw[i]; }
    }
#endif
    __syncthreads();
  }

  // epilogue: bias add + store (f32 or f16 output)
#pragma unroll
  for (int sm = 0; sm < 2; ++sm) {
#pragma unroll
    for (int sn = 0; sn < 4; ++sn) {
      const int n = n0 + wn * 64 + sn * 16 + colL;
      const float bv = bias[n];
#pragma unroll
      for (int j = 0; j < 8; ++j) {
        const int m = m0 + wm * 32 + sm * 16 + j + rowoff;
        out[(size_t)m * N + n] = (OutT)(acc[sm][sn][j] + bv);
      }
    }
  }
}

// ---------------------------------------------------------------------------
// Interleaved RoPE on (B,S,H,D_K) fp32, one thread per (even,odd) pair.
// ---------------------------------------------------------------------------
__global__ void rope_kernel(float* __restrict__ T, int n) {
  const int idx = blockIdx.x * blockDim.x + threadIdx.x;
  if (idx >= n) return;
  const int k = idx & 31;
  int rest = idx >> 5;
  const int h = rest & (N_HEADS - 1);
  rest >>= 4;
  const int s = rest & (SEQ - 1);
  const int b = rest >> 11;
  const float ang = (float)s * __powf(10000.0f, -(2.0f * (float)k) / (float)D_K);
  float sn, cs;
  __sincosf(ang, &sn, &cs);
  float* p = T + ((size_t)b * SEQ + s) * D_MODEL + h * D_K + 2 * k;
  const float x1 = p[0], x2 = p[1];
  p[0] = cs * x1 - sn * x2;
  p[1] = sn * x1 + cs * x2;
}

// ---------------------------------------------------------------------------
// Causal flash attention: grid (S/64, H, B), 128 threads = 4 waves.
// Q,K are f32 (post-RoPE); V is pre-converted f16; ctx written as f16.
// ---------------------------------------------------------------------------
__global__ __launch_bounds__(128) void attn_kernel(const float* __restrict__ Q,
                                                   const float* __restrict__ Kin,
                                                   const _Float16* __restrict__ Vin,
                                                   _Float16* __restrict__ Ctx) {
  __shared__ alignas(16) _Float16 Kt[32][64];      // [key][d]
  __shared__ alignas(16) _Float16 Vt[64][32];      // [d][key] (transposed)
  __shared__ alignas(16) _Float16 Ps[4][16][32];   // per-wave P tile [m][key]

  const int tid    = threadIdx.x;
  const int lane   = tid & 31;
  const int wave   = tid >> 5;
  const int col    = lane & 15;
  const int kbA    = (lane < 16) ? 0 : 8;
  const int kbB    = (lane < 16) ? 0 : 8;
  const int rowoff = (lane < 16) ? 0 : 8;

  const int q0   = blockIdx.x * 64;
  const int h    = blockIdx.y;
  const int b    = blockIdx.z;
  const int qrow = q0 + wave * 16;

  // Q fragments (scaled by 1/sqrt(d_k)=0.125, folded into the f16 convert)
  const float* qp = Q + ((size_t)b * SEQ + qrow + col) * D_MODEL + h * D_K;
  const v16h aq0 = frag16_from_f32_scl(qp + kbA, qp + 16 + kbA, 0.125f);        // d 0..31
  const v16h aq1 = frag16_from_f32_scl(qp + 32 + kbA, qp + 48 + kbA, 0.125f);   // d 32..63

  v8f zero = {};
  v8f acc[4] = {zero, zero, zero, zero};
  float mrow[8], lrow[8];
#pragma unroll
  for (int j = 0; j < 8; ++j) { mrow[j] = -1e30f; lrow[j] = 0.0f; }

  const int kend = q0 + 64;
  for (int kc = 0; kc < kend; kc += 32) {
    __syncthreads();
    for (int e = tid; e < 32 * 64; e += 128) {
      const int key = e >> 6, d = e & 63;
      const size_t g = ((size_t)b * SEQ + kc + key) * D_MODEL + h * D_K + d;
      Kt[key][d] = (_Float16)Kin[g];
      Vt[d][key] = Vin[g];
    }
    __syncthreads();

    v8f s0 = __builtin_amdgcn_wmma_f32_16x16x32_f16(false, aq0, false,
               frag16_lds(&Kt[col][kbB]),        (short)0, zero, false, false);
    s0     = __builtin_amdgcn_wmma_f32_16x16x32_f16(false, aq1, false,
               frag16_lds(&Kt[col][32 + kbB]),   (short)0, s0,   false, false);
    v8f s1 = __builtin_amdgcn_wmma_f32_16x16x32_f16(false, aq0, false,
               frag16_lds(&Kt[16 + col][kbB]),      (short)0, zero, false, false);
    s1     = __builtin_amdgcn_wmma_f32_16x16x32_f16(false, aq1, false,
               frag16_lds(&Kt[16 + col][32 + kbB]), (short)0, s1,   false, false);

    const int kg0 = kc + col, kg1 = kc + 16 + col;
    float p0[8], p1[8], scl[8];
#pragma unroll
    for (int j = 0; j < 8; ++j) {
      const int qg = qrow + j + rowoff;
      const float v0 = (kg0 <= qg) ? s0[j] : -1e30f;
      const float v1 = (kg1 <= qg) ? s1[j] : -1e30f;
      float rmx = fmaxf(v0, v1);
#pragma unroll
      for (int mk = 1; mk < 16; mk <<= 1) rmx = fmaxf(rmx, __shfl_xor(rmx, mk));
      const float mn = fmaxf(mrow[j], rmx);
      scl[j]  = __expf(mrow[j] - mn);
      mrow[j] = mn;
      p0[j] = __expf(v0 - mn);
      p1[j] = __expf(v1 - mn);
      float rs = p0[j] + p1[j];
#pragma unroll
      for (int mk = 1; mk < 16; mk <<= 1) rs += __shfl_xor(rs, mk);
      lrow[j] = lrow[j] * scl[j] + rs;
    }
#pragma unroll
    for (int dt = 0; dt < 4; ++dt)
#pragma unroll
      for (int j = 0; j < 8; ++j) acc[dt][j] *= scl[j];

#pragma unroll
    for (int j = 0; j < 8; ++j) {
      Ps[wave][j + rowoff][col]      = (_Float16)p0[j];
      Ps[wave][j + rowoff][16 + col] = (_Float16)p1[j];
    }
    __syncthreads();

    const v16h ap = frag_a_lds(&Ps[wave][col][0], kbA);
#pragma unroll
    for (int dt = 0; dt < 4; ++dt) {
      v16h bv = frag16_lds(&Vt[16 * dt + col][kbB]);
      acc[dt] = __builtin_amdgcn_wmma_f32_16x16x32_f16(false, ap, false, bv,
                                                       (short)0, acc[dt], false, false);
    }
  }

  float invl[8];
#pragma unroll
  for (int j = 0; j < 8; ++j) invl[j] = 1.0f / (lrow[j] + 1e-8f);
#pragma unroll
  for (int dt = 0; dt < 4; ++dt) {
#pragma unroll
    for (int j = 0; j < 8; ++j) {
      const int m = qrow + j + rowoff;
      const int d = 16 * dt + col;
      Ctx[((size_t)b * SEQ + m) * D_MODEL + h * D_K + d] = (_Float16)(acc[dt][j] * invl[j]);
    }
  }
}

// ---------------------------------------------------------------------------
extern "C" void kernel_launch(void* const* d_in, const int* in_sizes, int n_in,
                              void* d_out, int out_size, void* d_ws, size_t ws_size,
                              hipStream_t stream) {
  (void)in_sizes; (void)n_in; (void)out_size; (void)ws_size;
  const float* x  = (const float*)d_in[0];
  const float* Wq = (const float*)d_in[1];
  const float* bq = (const float*)d_in[2];
  const float* Wk = (const float*)d_in[3];
  const float* bk = (const float*)d_in[4];
  const float* Wv = (const float*)d_in[5];
  const float* bv = (const float*)d_in[6];
  const float* Wo = (const float*)d_in[7];
  const float* bo = (const float*)d_in[8];
  float* out = (float*)d_out;

  const size_t plane = (size_t)BATCH * SEQ * D_MODEL;   // 8.39M elems
  const size_t wsz   = (size_t)D_MODEL * D_MODEL;       // 1.05M elems

  // workspace layout (halfs first, then floats; everything 16B aligned)
  _Float16* xh   = (_Float16*)d_ws;     // plane halfs
  _Float16* wqh  = xh + plane;          // wsz
  _Float16* wkh  = wqh + wsz;
  _Float16* wvh  = wkh + wsz;
  _Float16* woh  = wvh + wsz;
  _Float16* vh   = woh + wsz;           // plane
  _Float16* ctxh = vh + plane;          // plane
  float*    Qf   = (float*)(ctxh + plane);   // plane f32
  float*    Kf   = Qf + plane;               // plane f32

  const int M = BATCH * SEQ;                        // 8192
  dim3 gg(D_MODEL / NT, M / MT);                    // (8, 64), 256 thr each

  // 1) one-pass f32->f16 of x and weights
  cvt_f32_f16<<<(int)(plane / 8 + 255) / 256, 256, 0, stream>>>(x, xh, (int)plane);
  cvt_f32_f16<<<(int)(wsz / 8 + 255) / 256, 256, 0, stream>>>(Wq, wqh, (int)wsz);
  cvt_f32_f16<<<(int)(wsz / 8 + 255) / 256, 256, 0, stream>>>(Wk, wkh, (int)wsz);
  cvt_f32_f16<<<(int)(wsz / 8 + 255) / 256, 256, 0, stream>>>(Wv, wvh, (int)wsz);
  cvt_f32_f16<<<(int)(wsz / 8 + 255) / 256, 256, 0, stream>>>(Wo, woh, (int)wsz);

  // 2) projections (V written directly as f16; Q,K stay f32 for RoPE)
  gemm_h<float>   <<<gg, 256, 0, stream>>>(xh, wqh, bq, Qf, M, D_MODEL, D_MODEL);
  gemm_h<float>   <<<gg, 256, 0, stream>>>(xh, wkh, bk, Kf, M, D_MODEL, D_MODEL);
  gemm_h<_Float16><<<gg, 256, 0, stream>>>(xh, wvh, bv, vh, M, D_MODEL, D_MODEL);

  // 3) RoPE on Q and K
  const int nrope = BATCH * SEQ * N_HEADS * (D_K / 2);
  rope_kernel<<<(nrope + 255) / 256, 256, 0, stream>>>(Qf, nrope);
  rope_kernel<<<(nrope + 255) / 256, 256, 0, stream>>>(Kf, nrope);

  // 4) causal attention -> f16 context
  attn_kernel<<<dim3(SEQ / 64, N_HEADS, BATCH), 128, 0, stream>>>(Qf, Kf, vh, ctxh);

  // 5) output projection
  gemm_h<float><<<gg, 256, 0, stream>>>(ctxh, woh, bo, out, M, D_MODEL, D_MODEL);
}
